// EGNN_15710990369456
// MI455X (gfx1250) — matrix operand order repaired and checked by reading, not compiled, not run on hardware
//
#include <hip/hip_runtime.h>
#include <hip/hip_bf16.h>
#include <math.h>

// ---------------------------------------------------------------------------
// EGNN forward for MI455X (gfx1250, wave32, WMMA).
// All GEMMs run on v_wmma_f32_16x16x32_f16 (f16 in, f32 accum).
// Workspace requirement: ~780 MB.
// ---------------------------------------------------------------------------

#define N_NODES 50000
#define N_EDGES 200000
#define NODE_IN 128
#define EDGE_IN 16
#define HID     256
#define OUT_DIM 256
#define NLAYERS 3

typedef __attribute__((ext_vector_type(16))) _Float16 v16h;
typedef __attribute__((ext_vector_type(8)))  _Float16 v8h;
typedef __attribute__((ext_vector_type(8)))  float    v8f;

#define GF_RELU    1
#define GF_F16     2
#define GF_F32     4
#define GF_SCATTER 8

// One concatenated-A segment's worth of K-loop. Addressing stays directly
// derived from the kernel-arg pointer (no dynamic pointer select), so the
// compiler's address-space inference emits global_load_b128 (not flat_load),
// keeping DScnt free and the LDS path idle.
__device__ __forceinline__
void gemm_seg(const _Float16* __restrict__ base, const int* __restrict__ idx,
              int Kseg, int kOff, int grc, int g,
              const _Float16* const __restrict__ Bp[4], v8f acc[4])
{
    int r = idx ? idx[grc] : grc;
    const _Float16* ap = base + (long)r * Kseg + 16 * g;
    for (int kl = 0; kl < Kseg; kl += 32) {
        // A frag (16x32 f16): lane(m16,g) holds K = 16g..16g+15 contiguous.
        v8h c1 = *(const v8h*)(ap + kl);
        v8h c2 = *(const v8h*)(ap + kl + 8);
        v16h a;
#pragma unroll
        for (int t = 0; t < 8; ++t) { a[t] = c1[t]; a[8 + t] = c2[t]; }
#pragma unroll
        for (int j = 0; j < 4; ++j) {
            // B frag (32x16 f16): lane n=m16, K = 16g..16g+15 contiguous in Bt.
            v16h b = *(const v16h*)(Bp[j] + kOff + kl);
            acc[j] = __builtin_amdgcn_wmma_f32_16x16x32_f16(
                false, a, false, b, (short)0, acc[j], false, false);
        }
    }
}

// ---------------------------------------------------------------------------
// Generic WMMA GEMM: C[M,N] = act( concat(A0[g0],A1[g1],A2[g2]) @ B + bias )
//  - A segments: f16, row stride == segment K, optional row-gather index.
//  - Bt: f16 weights pre-transposed to [N, Ktot] (so B frag loads are
//    contiguous per lane, same k order (16g..16g+15) as the A frag).
//  - Block = 128 threads = 4 waves; wave w computes rows [16w,16w+16) of a
//    64-row tile, 64 cols (4 accumulators). grid = (ceil(M/64), N/64).
//  - Epilogue: bias, relu, store f32 and/or f16, or atomic scatter-add
//    (fused segment_sum for the message pass).
// ---------------------------------------------------------------------------
__global__ __launch_bounds__(128)
void gemm_wmma_kernel(const _Float16* __restrict__ A0, const int* __restrict__ I0, int K0,
                      const _Float16* __restrict__ A1, const int* __restrict__ I1, int K1,
                      const _Float16* __restrict__ A2, const int* __restrict__ I2, int K2,
                      const _Float16* __restrict__ Bt, const float* __restrict__ bias,
                      float* __restrict__ Cf, _Float16* __restrict__ Ch,
                      const int* __restrict__ SIdx,
                      int M, int N, int flags)
{
    const int Ktot = K0 + K1 + K2;
    const int lane = threadIdx.x & 31;
    const int wave = threadIdx.x >> 5;
    const int m16  = lane & 15;       // row (A) / col (B,C) lane mapping
    const int g    = lane >> 4;       // lane group
    const int rowTile = blockIdx.x * 64 + wave * 16;
    const int col0    = blockIdx.y * 64;

    int gr  = rowTile + m16;
    int grc = (gr < M) ? gr : (M - 1);          // clamp loads, guard stores

    // B fragment row pointers (one 16-col group per accumulator).
    const _Float16* Bp[4];
#pragma unroll
    for (int j = 0; j < 4; ++j) {
        int n = col0 + 16 * j + m16;
        Bp[j] = Bt + (long)n * Ktot + 16 * g;
    }

    v8f acc[4];
#pragma unroll
    for (int j = 0; j < 4; ++j)
#pragma unroll
        for (int t = 0; t < 8; ++t) acc[j][t] = 0.0f;

    // Up to 3 concatenated A segments; uniform scalar branches keep EXEC full.
    gemm_seg(A0, I0, K0, 0, grc, g, Bp, acc);
    if (K1) gemm_seg(A1, I1, K1, K0, grc, g, Bp, acc);
    if (K2) gemm_seg(A2, I2, K2, K0 + K1, grc, g, Bp, acc);

    const bool relu = (flags & GF_RELU) != 0;
#pragma unroll
    for (int j = 0; j < 4; ++j) {
        int n = col0 + 16 * j + m16;
        float bv = bias ? bias[n] : 0.0f;
#pragma unroll
        for (int r = 0; r < 8; ++r) {
            int row = rowTile + 8 * g + r;   // C frag: vgpr r holds M = 8g + r
            if (row < M) {
                float v = acc[j][r] + bv;
                if (relu && v < 0.0f) v = 0.0f;
                if (flags & GF_SCATTER) {
                    atomicAdd(&Cf[(long)SIdx[row] * N + n], v);
                } else {
                    if (flags & GF_F32) Cf[(long)row * N + n] = v;
                    if (flags & GF_F16) Ch[(long)row * N + n] = (_Float16)v;
                }
            }
        }
    }
}

// ---------------------------------------------------------------------------
// Support kernels
// ---------------------------------------------------------------------------
#define GSTRIDE(i, n) for (long i = blockIdx.x * (long)blockDim.x + threadIdx.x; \
                           i < (n); i += (long)gridDim.x * blockDim.x)

// Convert + transpose f32 weight [K,N] -> f16 [N,Kpad] (zero-padded K).
__global__ void wt_kernel(const float* __restrict__ W, _Float16* __restrict__ Wt,
                          int K, int N, int Kpad)
{
    long total = (long)N * Kpad;
    GSTRIDE(i, total) {
        int k = (int)(i % Kpad);
        int n = (int)(i / Kpad);
        Wt[i] = (k < K) ? (_Float16)W[(long)k * N + n] : (_Float16)0.0f;
    }
}

__global__ void f2h_kernel(const float* __restrict__ in, _Float16* __restrict__ out, long n)
{
    GSTRIDE(i, n) out[i] = (_Float16)in[i];
}

// edge_attr [E,16] f32 -> [E,32] f16 zero-padded (so K=32 fits one WMMA step).
__global__ void pad_edge_kernel(const float* __restrict__ ea, _Float16* __restrict__ out, long n)
{
    GSTRIDE(i, n) {
        int  k = (int)(i & 31);
        long e = i >> 5;
        out[i] = (k < EDGE_IN) ? (_Float16)ea[e * EDGE_IN + k] : (_Float16)0.0f;
    }
}

// x = (addres ? x : 0) + relu(bn(xn)) ; also refresh f16 copy.
__global__ void bn_relu_res_kernel(const float* __restrict__ xn,
                                   const float* __restrict__ gamma,
                                   const float* __restrict__ beta,
                                   float* __restrict__ xout, _Float16* __restrict__ xh,
                                   long n, int addres)
{
    const float inv = 0.99999500003749968f;   // 1/sqrt(1 + 1e-5)
    GSTRIDE(i, n) {
        int c = (int)(i & (HID - 1));
        float v = xn[i] * (gamma[c] * inv) + beta[c];
        if (v < 0.0f) v = 0.0f;
        float o = addres ? (xout[i] + v) : v;
        xout[i] = o;
        xh[i]   = (_Float16)o;
    }
}

// Final Linear(128 -> 1) head: s = a . w + b ; mode 1 => sigmoid.
__global__ void dot_head_kernel(const _Float16* __restrict__ A, int ld,
                                const float* __restrict__ W, const float* __restrict__ b,
                                float* __restrict__ out, int M, int mode)
{
    int row = blockIdx.x * blockDim.x + threadIdx.x;
    if (row >= M) return;
    const _Float16* a = A + (long)row * ld;
    float s = b[0];
#pragma unroll 8
    for (int k = 0; k < 128; ++k) s += (float)a[k] * W[k];
    if (mode == 1) s = 1.0f / (1.0f + expf(-s));
    out[row] = s;
}

__global__ void reduce_max_kernel(const float* __restrict__ s, int n, float* __restrict__ red)
{
    __shared__ float sm[256];
    float m = -1e30f;
    for (int i = threadIdx.x; i < n; i += blockDim.x) m = fmaxf(m, s[i]);
    sm[threadIdx.x] = m; __syncthreads();
    for (int w = 128; w > 0; w >>= 1) {
        if (threadIdx.x < w) sm[threadIdx.x] = fmaxf(sm[threadIdx.x], sm[threadIdx.x + w]);
        __syncthreads();
    }
    if (threadIdx.x == 0) red[0] = sm[0];
}

__global__ void reduce_sumexp_kernel(const float* __restrict__ s, int n, float* __restrict__ red)
{
    __shared__ float sm[256];
    float mx = red[0];
    float a = 0.0f;
    for (int i = threadIdx.x; i < n; i += blockDim.x) a += expf(s[i] - mx);
    sm[threadIdx.x] = a; __syncthreads();
    for (int w = 128; w > 0; w >>= 1) {
        if (threadIdx.x < w) sm[threadIdx.x] += sm[threadIdx.x + w];
        __syncthreads();
    }
    if (threadIdx.x == 0) red[1] = sm[0];
}

__global__ void softmax_norm_kernel(const float* __restrict__ s, int n,
                                    const float* __restrict__ red, float* __restrict__ out)
{
    GSTRIDE(i, (long)n) out[i] = expf(s[i] - red[0]) / red[1];
}

// graph[c] = sum_i x[i,c] * attn[i]   (block-partial sums + one atomic/col)
__global__ void wsum_kernel(const float* __restrict__ x, const float* __restrict__ attn,
                            int n, float* __restrict__ acc)
{
    int col = threadIdx.x;                 // 256 threads = HID cols
    float s = 0.0f;
    for (int i = blockIdx.x; i < n; i += gridDim.x) s += x[(long)i * HID + col] * attn[i];
    atomicAdd(&acc[col], s);
}

__global__ void outproj_kernel(const float* __restrict__ g, const float* __restrict__ W,
                               const float* __restrict__ b, float* __restrict__ out)
{
    int n = threadIdx.x;                   // 256 threads
    float s = b[n];
    for (int k = 0; k < HID; ++k) s += g[k] * W[(long)k * OUT_DIM + n];
    out[n] = s;
}

// ---------------------------------------------------------------------------
// Host side
// ---------------------------------------------------------------------------
struct ParamPtrs {
    const float *x, *ea_in;
    const float *Wni, *bni, *Wei, *bei;
    const float *Wem0[3], *bem0[3], *Wem1[3], *bem1[3], *Wem2[3], *bem2[3];
    const float *Wnm0[3], *bnm0[3], *Wnm1[3], *bnm1[3];
    const float *Weu0[3], *beu0[3], *Weu1[3], *beu1[3];
    const float *bng[3], *bnb[3], *beg[3], *beb[3];
    const float *Wna0, *bna0, *Wna1, *bna1;
    const float *Wea0, *bea0, *Wea1, *bea1;
    const float *Wt0, *bt0, *Wt1, *bt1;
    const float *Wop, *bop;
    const int   *ei;
};

static void map_inputs(void* const* d_in, const int* in_sizes, ParamPtrs& p)
{
    auto F = [&](int i) { return (const float*)d_in[i]; };
    bool insertion = (in_sizes[0] == N_NODES * NODE_IN);   // x first => insertion order
    if (insertion) {
        p.x = F(0); p.ea_in = F(1);
        p.Wni = F(2); p.bni = F(3); p.Wei = F(4); p.bei = F(5);
        for (int i = 0; i < 3; ++i) {
            int b = 6 + 14 * i;
            p.Wem0[i]=F(b+0);  p.bem0[i]=F(b+1);  p.Wem1[i]=F(b+2);  p.bem1[i]=F(b+3);
            p.Wem2[i]=F(b+4);  p.bem2[i]=F(b+5);
            p.Wnm0[i]=F(b+6);  p.bnm0[i]=F(b+7);  p.Wnm1[i]=F(b+8);  p.bnm1[i]=F(b+9);
            p.Weu0[i]=F(b+10); p.beu0[i]=F(b+11); p.Weu1[i]=F(b+12); p.beu1[i]=F(b+13);
        }
        for (int i = 0; i < 3; ++i) {
            p.bng[i]=F(48+2*i); p.bnb[i]=F(49+2*i);
            p.beg[i]=F(54+2*i); p.beb[i]=F(55+2*i);
        }
        p.Wna0=F(60); p.bna0=F(61); p.Wna1=F(62); p.bna1=F(63);
        p.Wea0=F(64); p.bea0=F(65); p.Wea1=F(66); p.bea1=F(67);
        p.Wt0 =F(68); p.bt0 =F(69); p.Wt1 =F(70); p.bt1 =F(71);
        p.Wop =F(72); p.bop =F(73);
        p.ei  = (const int*)d_in[74];
    } else {                                               // alphabetical pytree order
        p.ea_in = F(0); p.ei = (const int*)d_in[1];
        for (int i = 0; i < 3; ++i) {
            p.beg[i]=F(2+2*i); p.beb[i]=F(3+2*i);
            p.bng[i]=F(8+2*i); p.bnb[i]=F(9+2*i);
        }
        for (int i = 0; i < 3; ++i) {
            int b = 14 + 14 * i;
            p.Wem0[i]=F(b+0);  p.bem0[i]=F(b+1);  p.Wem1[i]=F(b+2);  p.bem1[i]=F(b+3);
            p.Wem2[i]=F(b+4);  p.bem2[i]=F(b+5);
            p.Weu0[i]=F(b+6);  p.beu0[i]=F(b+7);  p.Weu1[i]=F(b+8);  p.beu1[i]=F(b+9);
            p.Wnm0[i]=F(b+10); p.bnm0[i]=F(b+11); p.Wnm1[i]=F(b+12); p.bnm1[i]=F(b+13);
        }
        p.Wea0=F(56); p.bea0=F(57); p.Wea1=F(58); p.bea1=F(59);
        p.Wei =F(60); p.bei =F(61);
        p.Wna0=F(62); p.bna0=F(63); p.Wna1=F(64); p.bna1=F(65);
        p.Wni =F(66); p.bni =F(67);
        p.Wop =F(68); p.bop =F(69);
        p.Wt0 =F(70); p.bt0 =F(71); p.Wt1 =F(72); p.bt1 =F(73);
        p.x   =F(74);
    }
}

extern "C" void kernel_launch(void* const* d_in, const int* in_sizes, int n_in,
                              void* d_out, int out_size, void* d_ws, size_t ws_size,
                              hipStream_t stream)
{
    (void)n_in; (void)out_size; (void)ws_size;
    ParamPtrs p;
    map_inputs(d_in, in_sizes, p);
    const int* src = p.ei;
    const int* dst = p.ei + N_EDGES;

    // ---- output slices (write final x / ea / heads straight into d_out) ----
    float* out    = (float*)d_out;
    float* graph  = out;                                    // [256]
    float* xout   = out + OUT_DIM;                          // [50000,256]
    float* eaout  = xout + (long)N_NODES * HID;             // [200000,256]
    float* nattn  = eaout + (long)N_EDGES * HID;            // [50000]
    float* eattn  = nattn + N_NODES;                        // [200000]
    float* taint  = eattn + N_EDGES;                        // [50000]

    // ---- workspace bump allocator ----
    char* wbase = (char*)d_ws;
    size_t off = 0;
    auto alloc = [&](size_t bytes) -> void* {
        off = (off + 255) & ~(size_t)255;
        void* ptr = wbase + off;
        off += bytes;
        return ptr;
    };
    auto allocH = [&](long n) { return (_Float16*)alloc((size_t)n * 2); };
    auto allocF = [&](long n) { return (float*)alloc((size_t)n * 4); };

    // f16 transposed weights
    _Float16* Wt_ni = allocH((long)HID * NODE_IN);
    _Float16* Wt_ei = allocH((long)HID * 32);
    _Float16 *Wt_eu0[3], *Wt_eu1[3], *Wt_em0[3], *Wt_em1[3], *Wt_em2[3], *Wt_nm0[3], *Wt_nm1[3];
    for (int i = 0; i < 3; ++i) {
        Wt_eu0[i] = allocH((long)HID * 768);
        Wt_eu1[i] = allocH((long)HID * 256);
        Wt_em0[i] = allocH((long)HID * 768);
        Wt_em1[i] = allocH((long)HID * 256);
        Wt_em2[i] = allocH((long)HID * 256);
        Wt_nm0[i] = allocH((long)HID * 512);
        Wt_nm1[i] = allocH((long)HID * 256);
    }
    _Float16* Wt_na0 = allocH((long)128 * 256);
    _Float16* Wt_ea0 = allocH((long)128 * 256);
    _Float16* Wt_t0  = allocH((long)128 * 256);

    // activations
    _Float16* xh_in  = allocH((long)N_NODES * NODE_IN);     // f16 input x
    _Float16* eh_in  = allocH((long)N_EDGES * 32);          // padded f16 edge_attr
    _Float16* xh     = allocH((long)N_NODES * HID);         // current x (f16)
    _Float16* eah    = allocH((long)N_EDGES * HID);         // current ea (f16)
    float*    enewF  = allocF((long)N_EDGES * HID);         // e_new pre-BN (f32)
    _Float16* enewH  = allocH((long)N_EDGES * HID);         // e_new (f16)
    _Float16* tmpEA  = allocH((long)N_EDGES * HID);         // edge scratch A
    _Float16* tmpEB  = allocH((long)N_EDGES * HID);         // edge scratch B
    _Float16* tmpNh  = allocH((long)N_NODES * HID);         // node scratch (f16)
    float*    xnew   = (float*)tmpEB;                       // alias: free when used
    float*    agg    = allocF((long)N_NODES * HID);
    _Float16* aggh   = allocH((long)N_NODES * HID);
    float*    nscore = allocF(N_NODES);
    float*    red    = allocF(8);
    float*    gacc   = allocF(OUT_DIM);

    const int EW = 256;                                      // elementwise threads
    const int EB = 4096;                                     // elementwise blocks

    auto gemm = [&](const _Float16* A0, const int* I0, int K0,
                    const _Float16* A1, const int* I1, int K1,
                    const _Float16* A2, const int* I2, int K2,
                    const _Float16* Bt, const float* bias,
                    float* Cf, _Float16* Ch, const int* SIdx,
                    int M, int N, int flags) {
        dim3 g((M + 63) / 64, N / 64);
        gemm_wmma_kernel<<<g, 128, 0, stream>>>(A0, I0, K0, A1, I1, K1, A2, I2, K2,
                                                Bt, bias, Cf, Ch, SIdx, M, N, flags);
    };
    auto prep = [&](const float* W, _Float16* Wt, int K, int N, int Kpad) {
        long total = (long)N * Kpad;
        wt_kernel<<<(int)((total + EW - 1) / EW), EW, 0, stream>>>(W, Wt, K, N, Kpad);
    };

    // ---- weight prep (f32 [K,N] -> f16 [N,K]) ----
    prep(p.Wni, Wt_ni, NODE_IN, HID, NODE_IN);
    prep(p.Wei, Wt_ei, EDGE_IN, HID, 32);
    for (int i = 0; i < 3; ++i) {
        prep(p.Weu0[i], Wt_eu0[i], 768, HID, 768);
        prep(p.Weu1[i], Wt_eu1[i], 256, HID, 256);
        prep(p.Wem0[i], Wt_em0[i], 768, HID, 768);
        prep(p.Wem1[i], Wt_em1[i], 256, HID, 256);
        prep(p.Wem2[i], Wt_em2[i], 256, HID, 256);
        prep(p.Wnm0[i], Wt_nm0[i], 512, HID, 512);
        prep(p.Wnm1[i], Wt_nm1[i], 256, HID, 256);
    }
    prep(p.Wna0, Wt_na0, 256, 128, 256);
    prep(p.Wea0, Wt_ea0, 256, 128, 256);
    prep(p.Wt0,  Wt_t0,  256, 128, 256);

    // ---- input conversion ----
    f2h_kernel<<<EB, EW, 0, stream>>>(p.x, xh_in, (long)N_NODES * NODE_IN);
    pad_edge_kernel<<<EB, EW, 0, stream>>>(p.ea_in, eh_in, (long)N_EDGES * 32);

    // ---- input projections ----
    gemm(xh_in, nullptr, NODE_IN, nullptr, nullptr, 0, nullptr, nullptr, 0,
         Wt_ni, p.bni, xout, xh, nullptr, N_NODES, HID, GF_RELU | GF_F32 | GF_F16);
    gemm(eh_in, nullptr, 32, nullptr, nullptr, 0, nullptr, nullptr, 0,
         Wt_ei, p.bei, eaout, eah, nullptr, N_EDGES, HID, GF_RELU | GF_F32 | GF_F16);

    // ---- conv layers ----
    for (int i = 0; i < NLAYERS; ++i) {
        // edge update: relu(L1([x_src, x_dst, ea])) -> L2 = e_new
        gemm(xh, src, 256, xh, dst, 256, eah, nullptr, 256,
             Wt_eu0[i], p.beu0[i], nullptr, tmpEA, nullptr, N_EDGES, HID, GF_RELU | GF_F16);
        gemm(tmpEA, nullptr, 256, nullptr, nullptr, 0, nullptr, nullptr, 0,
             Wt_eu1[i], p.beu1[i], enewF, enewH, nullptr, N_EDGES, HID, GF_F32 | GF_F16);
        // message MLP: relu(L1([x_dst, x_src, e_new])) -> relu(L2) -> L3, scatter-add at dst
        gemm(xh, dst, 256, xh, src, 256, enewH, nullptr, 256,
             Wt_em0[i], p.bem0[i], nullptr, tmpEB, nullptr, N_EDGES, HID, GF_RELU | GF_F16);
        gemm(tmpEB, nullptr, 256, nullptr, nullptr, 0, nullptr, nullptr, 0,
             Wt_em1[i], p.bem1[i], nullptr, tmpEA, nullptr, N_EDGES, HID, GF_RELU | GF_F16);
        hipMemsetAsync(agg, 0, (size_t)N_NODES * HID * 4, stream);
        gemm(tmpEA, nullptr, 256, nullptr, nullptr, 0, nullptr, nullptr, 0,
             Wt_em2[i], p.bem2[i], agg, nullptr, dst, N_EDGES, HID, GF_SCATTER);
        // node MLP: relu(L1([x, agg])) -> L2 = x_new
        f2h_kernel<<<EB, EW, 0, stream>>>(agg, aggh, (long)N_NODES * HID);
        gemm(xh, nullptr, 256, aggh, nullptr, 256, nullptr, nullptr, 0,
             Wt_nm0[i], p.bnm0[i], nullptr, tmpNh, nullptr, N_NODES, HID, GF_RELU | GF_F16);
        gemm(tmpNh, nullptr, 256, nullptr, nullptr, 0, nullptr, nullptr, 0,
             Wt_nm1[i], p.bnm1[i], xnew, nullptr, nullptr, N_NODES, HID, GF_F32);
        // BN + relu (+ residual for i>0); refresh f32 (d_out slice) and f16 copies
        bn_relu_res_kernel<<<EB, EW, 0, stream>>>(xnew, p.bng[i], p.bnb[i],
                                                  xout, xh, (long)N_NODES * HID, i > 0);
        bn_relu_res_kernel<<<EB, EW, 0, stream>>>(enewF, p.beg[i], p.beb[i],
                                                  eaout, eah, (long)N_EDGES * HID, i > 0);
    }

    // ---- node attention: softmax over nodes ----
    gemm(xh, nullptr, 256, nullptr, nullptr, 0, nullptr, nullptr, 0,
         Wt_na0, p.bna0, nullptr, tmpNh, nullptr, N_NODES, 128, GF_RELU | GF_F16);
    dot_head_kernel<<<(N_NODES + 255) / 256, 256, 0, stream>>>(tmpNh, 128, p.Wna1, p.bna1,
                                                               nscore, N_NODES, 0);
    reduce_max_kernel<<<1, 256, 0, stream>>>(nscore, N_NODES, red);
    reduce_sumexp_kernel<<<1, 256, 0, stream>>>(nscore, N_NODES, red);
    softmax_norm_kernel<<<EB, EW, 0, stream>>>(nscore, N_NODES, red, nattn);

    // ---- edge attention: sigmoid ----
    gemm(eah, nullptr, 256, nullptr, nullptr, 0, nullptr, nullptr, 0,
         Wt_ea0, p.bea0, nullptr, tmpEA, nullptr, N_EDGES, 128, GF_RELU | GF_F16);
    dot_head_kernel<<<(N_EDGES + 255) / 256, 256, 0, stream>>>(tmpEA, 128, p.Wea1, p.bea1,
                                                               eattn, N_EDGES, 1);

    // ---- taint head: sigmoid ----
    gemm(xh, nullptr, 256, nullptr, nullptr, 0, nullptr, nullptr, 0,
         Wt_t0, p.bt0, nullptr, tmpNh, nullptr, N_NODES, 128, GF_RELU | GF_F16);
    dot_head_kernel<<<(N_NODES + 255) / 256, 256, 0, stream>>>(tmpNh, 128, p.Wt1, p.bt1,
                                                               taint, N_NODES, 1);

    // ---- graph readout + output projection ----
    hipMemsetAsync(gacc, 0, OUT_DIM * 4, stream);
    wsum_kernel<<<256, 256, 0, stream>>>(xout, nattn, N_NODES, gacc);
    outproj_kernel<<<1, 256, 0, stream>>>(gacc, p.Wop, p.bop, graph);
}
